// ClusteringLayer_84593675862790
// MI455X (gfx1250) — matrix operand order repaired and checked by reading, not compiled, not run on hardware
//
#include <hip/hip_runtime.h>
#include <hip/hip_bf16.h>

// ---------------------------------------------------------------------------
// ClusteringLayer (Student-t soft assignment), MI455X / gfx1250, wave32.
//   N=65536 rows of x[D=512], K=256 clusters, fp32 in/out, ALPHA=1.
//   out[n,k] = q / rowsum(q),  q = 1/(1 + max(0, |x_n|^2 + |c_k|^2 - 2 x_n.c_k))
//
// GEMM path: split-bf16 (x = hi + lo) through v_wmma_f32_16x16x32_bf16,
// keeping hi*hi + hi*lo + lo*hi  (~fp32 accuracy, bf16 matrix-pipe speed).
// x tiles are staged into LDS by the Tensor Data Mover (TDM), double-buffered,
// with hardware row padding (32 DWORD rows + 4 DWORD pad = 36-float stride).
// ---------------------------------------------------------------------------

#define Nrows 65536
#define Dd    512
#define Kc    256

typedef __attribute__((ext_vector_type(16))) __bf16          v16bf;
typedef __attribute__((ext_vector_type(16))) unsigned short  v16u16;
typedef __attribute__((ext_vector_type(8)))  float           v8f;
typedef __attribute__((ext_vector_type(4)))  unsigned int    v4u;
typedef __attribute__((ext_vector_type(8)))  int             v8i;
typedef __attribute__((ext_vector_type(4)))  int             v4i;

#if __has_builtin(__builtin_amdgcn_tensor_load_to_lds) && \
    __has_builtin(__builtin_amdgcn_s_wait_tensorcnt)
#define USE_TDM 1
#else
#define USE_TDM 0
#endif

// round-to-nearest-even fp32 -> bf16 (bit trick, no __bf16 arithmetic needed)
__device__ __forceinline__ unsigned short f32_to_bf16(float v) {
    unsigned int u = __float_as_uint(v);
    u += 0x7FFFu + ((u >> 16) & 1u);
    return (unsigned short)(u >> 16);
}
__device__ __forceinline__ float bf16_to_f32(unsigned short h) {
    return __uint_as_float(((unsigned int)h) << 16);
}

// ---------------------------------------------------------------------------
// Prep: clusters[256x512] f32 -> bf16 hi/lo planes (row-major) + c_sq[256].
// 512 KB of bf16 planes -> lives in the 192 MB L2 for the whole main kernel.
// ---------------------------------------------------------------------------
__global__ __launch_bounds__(128) void prep_clusters(
    const float* __restrict__ clusters,
    unsigned short* __restrict__ chi,
    unsigned short* __restrict__ clo,
    float* __restrict__ c_sq)
{
    const int k = blockIdx.x;          // cluster row
    const int t = threadIdx.x;         // 128 threads
    float s = 0.0f;
    for (int i = t; i < Dd; i += 128) {
        float v = clusters[k * Dd + i];
        unsigned short h = f32_to_bf16(v);
        float lo = v - bf16_to_f32(h);
        chi[k * Dd + i] = h;
        clo[k * Dd + i] = f32_to_bf16(lo);
        s += v * v;
    }
    // wave reduce (wave32)
    for (int m = 16; m >= 1; m >>= 1) s += __shfl_xor(s, m, 32);
    __shared__ float part[4];
    const int lane = t & 31, w = t >> 5;
    if (lane == 0) part[w] = s;
    __syncthreads();
    if (t == 0) c_sq[k] = part[0] + part[1] + part[2] + part[3];
}

// ---------------------------------------------------------------------------
// Main: each block = 128 rows x all 256 columns (full row -> normalize here).
// 8 waves; wave w owns rows [blk*128 + w*16, +16) and 16 accum tiles (v8f).
// ---------------------------------------------------------------------------
#define ROWS_PER_BLOCK 128
#define LDS_STRIDE     36   // 32-float TDM row + 4-float HW pad (bank spread)

#if USE_TDM
// Issue one TDM 2-D tile load: 128 rows x 32 f32 from global (row stride 512
// elems) into LDS at lds_off, padding +4 DWORDs after every 32-DWORD row.
// Per-wave instruction (EXEC ignored); tracked with TENSORcnt.
// This toolchain exposes the 6-operand builtin:
//   (uint32x4 g0, int32x8 g1, int32x4 g2, int32x4 g3, int32x8 gx, i32 cpol)
__device__ __forceinline__ void tdm_load_tile(unsigned int lds_off,
                                              const float* gsrc) {
    const unsigned long long ga = (unsigned long long)(uintptr_t)gsrc;
    v4u g0;
    g0[0] = 1u;                                        // count=1, user D#
    g0[1] = lds_off;                                   // LDS byte address
    g0[2] = (unsigned int)ga;                          // global_addr[31:0]
    g0[3] = (unsigned int)((ga >> 32) & 0x01FFFFFFu)   // global_addr[56:32]
          | (2u << 30);                                // type = 2 ("image")
    v8i g1;
    g1[0] = (int)((2u << 16)      // data_size = 4 bytes
          |      (1u << 20)      // pad_enable
          |      (4u << 22)      // pad_interval: 2^(4+1)=32 DWORDs
          |      (3u << 25));    // pad_amount:   3+1 = 4 DWORDs
    g1[1] = (int)(512u << 16);   // tensor_dim0 = 512 (bits[79:48] lo half)
    g1[2] = 0;                   // tensor_dim0 hi=0, tensor_dim1 lo16(65536)=0
    g1[3] = (int)(1u             // tensor_dim1 hi16 -> 65536 rows
          | (32u << 16));        // tile_dim0 = 32 elements
    g1[4] = 128;                 // tile_dim1 = 128 rows (tile_dim2 = 0)
    g1[5] = 512;                 // tensor_dim0_stride = 512 elements (lo 32)
    g1[6] = 0;                   // stride hi, tensor_dim1_stride lo
    g1[7] = 0;
    const v4i z4 = {0, 0, 0, 0};            // groups 2/3 unused (2-D tile)
    const v8i z8 = {0, 0, 0, 0, 0, 0, 0, 0};
    __builtin_amdgcn_tensor_load_to_lds(g0, g1, z4, z4, z8, 0);
}
#endif

__global__ __launch_bounds__(256) void tdist_assign(
    const float* __restrict__ x,
    const unsigned short* __restrict__ chi,
    const unsigned short* __restrict__ clo,
    const float* __restrict__ c_sq,
    float* __restrict__ out)
{
    __shared__ float xs[2][ROWS_PER_BLOCK * LDS_STRIDE]; // double-buffered tile
    __shared__ float xsq_s[ROWS_PER_BLOCK];              // |x_row|^2 per row

    const int tid  = threadIdx.x;
    const int lane = tid & 31;
    const int wave = tid >> 5;
    const int row0 = blockIdx.x * ROWS_PER_BLOCK;     // first x row of block
    const int m    = lane & 15;                       // A-frag row within tile
    const int half = lane >> 4;                       // lane half (K split)
    const int koff = half * 8;                        // A: K 0-15 vs 16-31 (at 2*koff)

    v8f acc[16];
    #pragma unroll
    for (int ct = 0; ct < 16; ++ct) {
        #pragma unroll
        for (int r = 0; r < 8; ++r) acc[ct][r] = 0.0f;
    }

    float xsq = 0.0f;   // this lane's partial sum of squares of its A chunks

    const float* xbase = x + (size_t)row0 * Dd;
#if USE_TDM
    const unsigned int lds0 = (unsigned int)(uintptr_t)(&xs[0][0]);
    const unsigned int lds1 = (unsigned int)(uintptr_t)(&xs[1][0]);
    if (wave == 0) tdm_load_tile(lds0, xbase);        // prime chunk 0
#endif

    for (int it = 0; it < 16; ++it) {                 // 16 D-chunks of 32
        const int d0 = it * 32;
        __syncthreads();  // everyone done reading the buffer we refill next
#if USE_TDM
        if (wave == 0) {
            if (it + 1 < 16) {
                // prefetch chunk it+1 into the other buffer, then make sure
                // chunk it (issued one TDM earlier, in-order) has landed
                tdm_load_tile((it & 1) ? lds0 : lds1, xbase + (d0 + 32));
                __builtin_amdgcn_s_wait_tensorcnt(1);
            } else {
                __builtin_amdgcn_s_wait_tensorcnt(0);
            }
        }
#else
        { // fallback staging: thread t -> 16 floats of the 128x32 tile
            const int srow  = tid >> 1;
            const int scol  = (tid & 1) * 16;
            const float4* g = (const float4*)(xbase + (size_t)srow * Dd + d0 + scol);
            float4* l       = (float4*)(&xs[it & 1][0] + srow * LDS_STRIDE + scol);
            #pragma unroll
            for (int q = 0; q < 4; ++q) l[q] = g[q];
        }
#endif
        __syncthreads();  // chunk `it` visible in xs[it&1]

        // ---- build A fragments (16x32 bf16, contiguous K per lane half) + hi/lo split
        v16u16 ahu, alu;
        {
            const float* xr = &xs[it & 1][0] + (wave * 16 + m) * LDS_STRIDE;
            #pragma unroll
            for (int i = 0; i < 8; ++i) {
                float v = xr[2 * koff + i];             // K = 16*half + i
                unsigned short h = f32_to_bf16(v);
                ahu[i] = h;
                alu[i] = f32_to_bf16(v - bf16_to_f32(h));
                xsq += v * v;
            }
            #pragma unroll
            for (int i = 0; i < 8; ++i) {
                float v = xr[2 * koff + 8 + i];         // K = 16*half + 8 + i
                unsigned short h = f32_to_bf16(v);
                ahu[8 + i] = h;
                alu[8 + i] = f32_to_bf16(v - bf16_to_f32(h));
                xsq += v * v;
            }
        }
        const v16bf ah = __builtin_bit_cast(v16bf, ahu);
        const v16bf al = __builtin_bit_cast(v16bf, alu);

        // ---- sweep all 16 column tiles; B straight from L2-resident planes
        const int dof = d0 + half * 16;                 // B: K half per lane
        #pragma unroll
        for (int ct = 0; ct < 16; ++ct) {
            const int c = ct * 16 + m;                  // cluster column
            const size_t boff = (size_t)c * Dd + dof;   // 32B contiguous
            v16bf bh = __builtin_bit_cast(v16bf, *(const v16u16*)(chi + boff));
            v16bf bl = __builtin_bit_cast(v16bf, *(const v16u16*)(clo + boff));
            // dot += hi*hi + lo*hi + hi*lo   (drop lo*lo, ~2^-16 relative)
            acc[ct] = __builtin_amdgcn_wmma_f32_16x16x32_bf16(
                false, ah, false, bh, (short)0, acc[ct], false, false);
            acc[ct] = __builtin_amdgcn_wmma_f32_16x16x32_bf16(
                false, al, false, bh, (short)0, acc[ct], false, false);
            acc[ct] = __builtin_amdgcn_wmma_f32_16x16x32_bf16(
                false, ah, false, bl, (short)0, acc[ct], false, false);
        }
    }

    // ---- finish |x|^2: merge lane halves, publish per-row values to LDS
    xsq += __shfl_xor(xsq, 16, 32);
    __syncthreads();
    if (lane < 16) xsq_s[wave * 16 + lane] = xsq;
    __syncthreads();

    // per-lane column squared norms (column = ct*16 + m, fixed per lane)
    float cs[16];
    #pragma unroll
    for (int ct = 0; ct < 16; ++ct) cs[ct] = c_sq[ct * 16 + m];

    // ---- epilogue: q = 1/(1+d^2), row-normalize across K=256, store
    #pragma unroll
    for (int r = 0; r < 8; ++r) {
        const int mrow = r + 8 * half;                       // row in wave tile
        const float xq = xsq_s[wave * 16 + mrow];
        float q[16];
        float rs = 0.0f;
        #pragma unroll
        for (int ct = 0; ct < 16; ++ct) {
            float d2 = fmaxf(xq + cs[ct] - 2.0f * acc[ct][r], 0.0f);
            float qq = 1.0f / (1.0f + d2);                   // ALPHA=1 path
            q[ct] = qq;
            rs += qq;
        }
        // sum across the 16 lanes of this half (covers all 256 columns)
        rs += __shfl_xor(rs, 1, 32);
        rs += __shfl_xor(rs, 2, 32);
        rs += __shfl_xor(rs, 4, 32);
        rs += __shfl_xor(rs, 8, 32);
        const float inv = 1.0f / rs;
        float* orow = out + (size_t)(row0 + wave * 16 + mrow) * Kc;
        #pragma unroll
        for (int ct = 0; ct < 16; ++ct)
            orow[ct * 16 + m] = q[ct] * inv;                 // coalesced 64B/half
    }
}

// ---------------------------------------------------------------------------
extern "C" void kernel_launch(void* const* d_in, const int* in_sizes, int n_in,
                              void* d_out, int out_size, void* d_ws, size_t ws_size,
                              hipStream_t stream) {
    const float* x        = (const float*)d_in[0];   // [N, D] f32
    const float* clusters = (const float*)d_in[1];   // [K, D] f32
    float* out            = (float*)d_out;           // [N, K] f32

    // workspace: bf16 hi plane | bf16 lo plane | c_sq   (~513 KB total)
    unsigned short* chi = (unsigned short*)d_ws;
    unsigned short* clo = chi + (size_t)Kc * Dd;
    float*          csq = (float*)(clo + (size_t)Kc * Dd);

    prep_clusters<<<Kc, 128, 0, stream>>>(clusters, chi, clo, csq);
    tdist_assign<<<Nrows / ROWS_PER_BLOCK, 256, 0, stream>>>(x, chi, clo, csq, out);
}
